// ConformerEncoderLayer_21620865368377
// MI455X (gfx1250) — compile-verified, matrix operand order checked by reading, not run on hardware
//
#include <hip/hip_runtime.h>

// ---------------- types ----------------
typedef _Float16 half_t;
typedef __attribute__((ext_vector_type(16))) _Float16 v16h;
typedef __attribute__((ext_vector_type(8)))  _Float16 h8;
typedef __attribute__((ext_vector_type(8)))  float    v8f;

// ---------------- problem dims ----------------
#define B_   4
#define T_   2048
#define D_   256
#define H_   4
#define DK_  64
#define DFF_ 2048
#define KW_  31
#define PAD_ 15
#define NT_  (B_*T_)
#define BTD_ (NT_*D_)

// ---------------- gemm tiling ----------------
#define BM   64
#define BN   64
#define BK   32
#define LDSA 40   // padded half-stride: 80B rows -> 20-bank stride, conflict free

// ============================================================
// fragment load: lane holds 16 f16 along K.
// lanes 0-15: k in [hs,hs+8) u [16+hs,16+hs+8), hs=(lane>>4)*8
// Two 16B ds_load_b128 per fragment.
// ============================================================
__device__ __forceinline__ v16h load_frag(const half_t* p) {
  h8 a = *(const h8*)(p);
  h8 b = *(const h8*)(p + 16);
  return __builtin_shufflevector(a, b, 0,1,2,3,4,5,6,7,8,9,10,11,12,13,14,15);
}

// straight-line epilogue, fully resolved at compile time
template<bool HAS_BIAS, int ACT, bool HAS_RES, bool OUT_F, bool OUT_H>
__device__ __forceinline__ void store_tile(
    v8f acc, int r0, int c, int ldc,
    const float* __restrict__ bias, const float* __restrict__ scale,
    const float* __restrict__ shiftp, const float* __restrict__ residual,
    float alpha, float* __restrict__ outf, half_t* __restrict__ outh)
{
  float bv = 0.0f, sv = 1.0f, tv = 0.0f;
  if constexpr (HAS_BIAS) bv = bias[c];
  if constexpr (ACT == 2) { sv = scale[c]; tv = shiftp[c]; }
#pragma unroll
  for (int e = 0; e < 8; ++e) {
    float v = acc[e] + bv;
    if constexpr (ACT == 2) v = v * sv + tv;
    if constexpr (ACT != 0) v = v / (1.0f + __expf(-v));   // swish
    size_t off = (size_t)(r0 + e) * (size_t)ldc + (size_t)c;
    if constexpr (HAS_RES) v = residual[off] + alpha * v;
    if constexpr (OUT_F) outf[off] = v;
    if constexpr (OUT_H) outh[off] = (half_t)v;
  }
}

// ============================================================
// C[M,N] = act(A[M,K] * B[K,N] + bias) (+ residual)  -- A,B f16
// grid = (N/64, M/64), block = 256 (8 waves, 2 WMMA tiles each)
// ============================================================
template<bool HAS_BIAS, int ACT, bool HAS_RES, bool OUT_F, bool OUT_H>
__global__ __launch_bounds__(256) void gemm_nn(
    const half_t* __restrict__ A, int lda,
    const half_t* __restrict__ Bm, int ldb, int Kdim,
    const float* __restrict__ bias,
    const float* __restrict__ scale, const float* __restrict__ shiftp,
    const float* __restrict__ residual, float alpha,
    float* __restrict__ outf, int ldc, half_t* __restrict__ outh)
{
  __shared__ __align__(16) half_t As[BM * LDSA];
  __shared__ __align__(16) half_t Bs[BN * LDSA];
  const int tid  = threadIdx.x;
  const int lane = tid & 31;
  const int wave = tid >> 5;
  const int row0 = blockIdx.y * BM;
  const int col0 = blockIdx.x * BN;
  const int mi   = wave >> 1;
  const int nb   = (wave & 1) * 32;

  v8f acc0 = {0.f,0.f,0.f,0.f,0.f,0.f,0.f,0.f};
  v8f acc1 = {0.f,0.f,0.f,0.f,0.f,0.f,0.f,0.f};

  const int am = tid >> 2;            // A: 0..63 row
  const int ak = (tid & 3) * 8;       // A: 16B segment
  const int bk = tid >> 3;            // B: 0..31 k-row
  const int bn = (tid & 7) * 8;       // B: 8-col segment

  for (int k0 = 0; k0 < Kdim; k0 += BK) {
    // A tile: 64x32 f16, one 16B load+store per thread
    *(float4*)(&As[am * LDSA + ak]) =
        *(const float4*)(A + (size_t)(row0 + am) * lda + (k0 + ak));
    // B tile: one 16B global load per thread, transpose-scatter to Bs[n][k]
    half_t tmp[8];
    *(float4*)tmp = *(const float4*)(Bm + (size_t)(k0 + bk) * ldb + (col0 + bn));
#pragma unroll
    for (int i = 0; i < 8; ++i) Bs[(bn + i) * LDSA + bk] = tmp[i];
    __syncthreads();

    const int hs = (lane >> 4) << 3;
    v16h afrag = load_frag(&As[(mi * 16 + (lane & 15)) * LDSA + hs]);
    v16h bf0   = load_frag(&Bs[(nb      + (lane & 15)) * LDSA + hs]);
    v16h bf1   = load_frag(&Bs[(nb + 16 + (lane & 15)) * LDSA + hs]);

    acc0 = __builtin_amdgcn_wmma_f32_16x16x32_f16(false, afrag, false, bf0, (short)0, acc0, false, false);
    acc1 = __builtin_amdgcn_wmma_f32_16x16x32_f16(false, afrag, false, bf1, (short)0, acc1, false, false);
    __syncthreads();
  }

  const int r0 = row0 + mi * 16 + ((lane >> 4) << 3);
  const int cg = col0 + nb + (lane & 15);
  store_tile<HAS_BIAS,ACT,HAS_RES,OUT_F,OUT_H>(acc0, r0, cg,      ldc, bias, scale, shiftp, residual, alpha, outf, outh);
  store_tile<HAS_BIAS,ACT,HAS_RES,OUT_F,OUT_H>(acc1, r0, cg + 16, ldc, bias, scale, shiftp, residual, alpha, outf, outh);
}

// ============================================================
// C[M,N] = A[M,K] * B[N,K]^T  (QK^T / QP^T) -- tiles staged with the
// CDNA5 async DMA path: GLOBAL_LOAD_ASYNC_TO_LDS_B128 (512B/wave/instr),
// synchronized with s_wait_asynccnt before the workgroup barrier.
// LDS byte offsets are taken from the real shared pointer (low 32 bits of
// the generic address) so the LDS global's address escapes into the asm
// and the fragment ds_loads cannot be folded away.
// ============================================================
__global__ __launch_bounds__(256) void gemm_nt(
    const half_t* __restrict__ A, int lda,
    const half_t* __restrict__ Bm, int ldb,
    int Kdim, float* __restrict__ outf, int ldc)
{
  __shared__ __align__(16) half_t smem[(BM + BN) * LDSA];
  half_t* As = smem;
  half_t* Bs = smem + BM * LDSA;

  const int tid  = threadIdx.x;
  const int lane = tid & 31;
  const int wave = tid >> 5;
  const int row0 = blockIdx.y * BM;
  const int col0 = blockIdx.x * BN;
  const int mi   = wave >> 1;
  const int nb   = (wave & 1) * 32;

  v8f acc0 = {0.f,0.f,0.f,0.f,0.f,0.f,0.f,0.f};
  v8f acc1 = {0.f,0.f,0.f,0.f,0.f,0.f,0.f,0.f};

  // per-thread 16B chunk: row (0..63), 8-half segment (0..3)
  const int crow = tid >> 2;
  const int cseg = (tid & 3) * 8;
  // LDS byte address from the generic pointer (aperture low bits = LDS offset)
  const unsigned ldsA = (unsigned)(uintptr_t)(&As[crow * LDSA + cseg]);
  const unsigned ldsB = (unsigned)(uintptr_t)(&Bs[crow * LDSA + cseg]);

  for (int k0 = 0; k0 < Kdim; k0 += BK) {
    const half_t* ga = A  + (size_t)(row0 + crow) * lda + (k0 + cseg);
    const half_t* gb = Bm + (size_t)(col0 + crow) * ldb + (k0 + cseg);
    // %2 pins the LDS array address as an escaped asm operand
    asm volatile("global_load_async_to_lds_b128 %0, %1, off"
                 :: "v"(ldsA), "v"(ga), "v"(smem) : "memory");
    asm volatile("global_load_async_to_lds_b128 %0, %1, off"
                 :: "v"(ldsB), "v"(gb), "v"(smem) : "memory");
    asm volatile("s_wait_asynccnt 0" ::: "memory");
    __syncthreads();

    const int hs = (lane >> 4) << 3;
    v16h afrag = load_frag(&As[(mi * 16 + (lane & 15)) * LDSA + hs]);
    v16h bf0   = load_frag(&Bs[(nb      + (lane & 15)) * LDSA + hs]);
    v16h bf1   = load_frag(&Bs[(nb + 16 + (lane & 15)) * LDSA + hs]);

    acc0 = __builtin_amdgcn_wmma_f32_16x16x32_f16(false, afrag, false, bf0, (short)0, acc0, false, false);
    acc1 = __builtin_amdgcn_wmma_f32_16x16x32_f16(false, afrag, false, bf1, (short)0, acc1, false, false);
    __syncthreads();
  }

  const int r0 = row0 + mi * 16 + ((lane >> 4) << 3);
  const int cg = col0 + nb + (lane & 15);
#pragma unroll
  for (int e = 0; e < 8; ++e) {
    outf[(size_t)(r0 + e) * ldc + cg]      = acc0[e];
    outf[(size_t)(r0 + e) * ldc + cg + 16] = acc1[e];
  }
}

// ============================================================
// LayerNorm over D=256 (one row / 256-thread block), dual f32+f16 out
// ============================================================
__global__ __launch_bounds__(256) void ln_kernel(
    const float* __restrict__ in, const float* __restrict__ g,
    const float* __restrict__ bt, float* __restrict__ outf,
    half_t* __restrict__ outh)
{
  __shared__ float red[256];
  const int r = blockIdx.x, t = threadIdx.x;
  float v = in[(size_t)r * D_ + t];
  red[t] = v; __syncthreads();
  for (int s = 128; s > 0; s >>= 1) { if (t < s) red[t] += red[t + s]; __syncthreads(); }
  float mean = red[0] * (1.0f / D_); __syncthreads();
  float d = v - mean;
  red[t] = d * d; __syncthreads();
  for (int s = 128; s > 0; s >>= 1) { if (t < s) red[t] += red[t + s]; __syncthreads(); }
  float var = red[0] * (1.0f / D_);
  float y = d * rsqrtf(var + 1e-5f) * g[t] + bt[t];
  outf[(size_t)r * D_ + t] = y;
  outh[(size_t)r * D_ + t] = (half_t)y;
}

// ---------------- elementwise helpers ----------------
__global__ __launch_bounds__(256) void cvt_f32_f16(
    const float* __restrict__ in, half_t* __restrict__ out, int n)
{
  int i = blockIdx.x * 256 + threadIdx.x;
  if (i < n) out[i] = (half_t)in[i];
}

// q,k,v [B,T,H,DK] f32 -> qu/qv/kf/vf [B,H,T,DK] f16 (+pos bias on q)
__global__ __launch_bounds__(256) void split_heads(
    const float* __restrict__ q, const float* __restrict__ k,
    const float* __restrict__ v, const float* __restrict__ pu,
    const float* __restrict__ pv, half_t* __restrict__ qu,
    half_t* __restrict__ qv, half_t* __restrict__ kf, half_t* __restrict__ vf)
{
  int idx = blockIdx.x * 256 + threadIdx.x;            // < B*T*D
  int dk = idx & (DK_ - 1);
  int h  = (idx >> 6) & (H_ - 1);
  int t  = (idx >> 8) & (T_ - 1);
  int b  = idx >> 19;
  size_t dst = ((size_t)((b * H_ + h) * T_ + t)) * DK_ + dk;
  float qq = q[idx];
  qu[dst] = (half_t)(qq + pu[h * DK_ + dk]);
  qv[dst] = (half_t)(qq + pv[h * DK_ + dk]);
  kf[dst] = (half_t)k[idx];
  vf[dst] = (half_t)v[idx];
}

__global__ __launch_bounds__(256) void split_p(
    const float* __restrict__ pfull, half_t* __restrict__ pf)
{
  int idx = blockIdx.x * 256 + threadIdx.x;            // < T*D
  int dk = idx & (DK_ - 1);
  int h  = (idx >> 6) & (H_ - 1);
  int t  = idx >> 8;
  pf[((size_t)(h * T_ + t)) * DK_ + dk] = (half_t)pfull[idx];
}

// ============================================================
// rel-shift + scale + mask + softmax (one query row per block)
// shifted_bd[i][j] = (jj==0) ? 0 : bdraw[qq][jj-1],
//   n=(i+1)*T+j, qq=n/(T+1), jj=n%(T+1)
// ============================================================
__global__ __launch_bounds__(256) void shift_softmax(
    float* __restrict__ attn, const float* __restrict__ bdraw,
    const int* __restrict__ mask, half_t* __restrict__ attnh)
{
  __shared__ float red[256];
  const int i = blockIdx.x, tid = threadIdx.x;
  float vals[8];
  float mx = -1e30f;
#pragma unroll
  for (int u = 0; u < 8; ++u) {
    int j  = tid + u * 256;
    int n  = (i + 1) * T_ + j;
    int qq = n / (T_ + 1);
    int jj = n - qq * (T_ + 1);
    float bd = (jj == 0) ? 0.0f : bdraw[(size_t)qq * T_ + (jj - 1)];
    float s  = (attn[(size_t)i * T_ + j] + bd) * 0.125f;   // 1/sqrt(64)
    if (mask[j] == 0) s = -10000.0f;
    vals[u] = s;
    mx = fmaxf(mx, s);
  }
  red[tid] = mx; __syncthreads();
  for (int s = 128; s > 0; s >>= 1) { if (tid < s) red[tid] = fmaxf(red[tid], red[tid + s]); __syncthreads(); }
  mx = red[0]; __syncthreads();
  float sum = 0.0f;
#pragma unroll
  for (int u = 0; u < 8; ++u) { vals[u] = __expf(vals[u] - mx); sum += vals[u]; }
  red[tid] = sum; __syncthreads();
  for (int s = 128; s > 0; s >>= 1) { if (tid < s) red[tid] += red[tid + s]; __syncthreads(); }
  float inv = 1.0f / red[0];
#pragma unroll
  for (int u = 0; u < 8; ++u) {
    int j = tid + u * 256;
    float a = vals[u] * inv;
    attn [(size_t)i * T_ + j] = a;
    attnh[(size_t)i * T_ + j] = (half_t)a;
  }
}

// GLU: c1[:, :D] * sigmoid(c1[:, D:])
__global__ __launch_bounds__(256) void glu_kernel(
    const float* __restrict__ c1, float* __restrict__ glu)
{
  int idx = blockIdx.x * 256 + threadIdx.x;  // < B*T*D
  int d  = idx & (D_ - 1);
  int bt = idx >> 8;
  float a = c1[(size_t)bt * (2 * D_) + d];
  float g = c1[(size_t)bt * (2 * D_) + D_ + d];
  glu[idx] = a / (1.0f + __expf(-g));
}

// depthwise conv1d K=31, pad 15 (channels-last, coalesced over d)
__global__ __launch_bounds__(256) void dwconv_kernel(
    const float* __restrict__ glu, const float* __restrict__ w,
    const float* __restrict__ bias, half_t* __restrict__ outh)
{
  int idx = blockIdx.x * 256 + threadIdx.x;  // < B*T*D
  int d = idx & (D_ - 1);
  int t = (idx >> 8) & (T_ - 1);
  int b = idx >> 19;
  float acc = bias[d];
#pragma unroll
  for (int k = 0; k < KW_; ++k) {
    int ti = t + k - PAD_;
    if (ti >= 0 && ti < T_)
      acc += w[d * KW_ + k] * glu[((size_t)(b * T_ + ti)) * D_ + d];
  }
  outh[idx] = (half_t)acc;
}

// fold BN (eval) + dc bias into per-column affine
__global__ __launch_bounds__(256) void bnprep_kernel(
    const float* g, const float* bb, const float* m, const float* v,
    const float* dcb, float* scale, float* shiftp)
{
  int c = threadIdx.x;
  float s = rsqrtf(v[c] + 1e-5f) * g[c];
  scale[c]  = s;
  shiftp[c] = (dcb[c] - m[c]) * s + bb[c];
}

// ============================================================
// host orchestration
// ============================================================
extern "C" void kernel_launch(void* const* d_in, const int* in_sizes, int n_in,
                              void* d_out, int out_size, void* d_ws, size_t ws_size,
                              hipStream_t stream) {
  (void)in_sizes; (void)n_in; (void)out_size; (void)ws_size;
  const float* x_in   = (const float*)d_in[0];
  const float* pe     = (const float*)d_in[1];
  const int*   mask   = (const int*)  d_in[2];
  const float* f1lg   = (const float*)d_in[3];
  const float* f1lb   = (const float*)d_in[4];
  const float* f1w1   = (const float*)d_in[5];
  const float* f1b1   = (const float*)d_in[6];
  const float* f1w2   = (const float*)d_in[7];
  const float* f1b2   = (const float*)d_in[8];
  const float* nlg    = (const float*)d_in[9];
  const float* nlb    = (const float*)d_in[10];
  const float* wq     = (const float*)d_in[11];
  const float* bq     = (const float*)d_in[12];
  const float* wk     = (const float*)d_in[13];
  const float* bk     = (const float*)d_in[14];
  const float* wv     = (const float*)d_in[15];
  const float* bv     = (const float*)d_in[16];
  const float* wpos   = (const float*)d_in[17];
  const float* pos_u  = (const float*)d_in[18];
  const float* pos_v  = (const float*)d_in[19];
  const float* wo     = (const float*)d_in[20];
  const float* bo     = (const float*)d_in[21];
  const float* cvlg   = (const float*)d_in[22];
  const float* cvlb   = (const float*)d_in[23];
  const float* pw1w   = (const float*)d_in[24];
  const float* pw1b   = (const float*)d_in[25];
  const float* dww    = (const float*)d_in[26];
  const float* dwb    = (const float*)d_in[27];
  const float* dcw    = (const float*)d_in[28];
  const float* dcb    = (const float*)d_in[29];
  const float* bng    = (const float*)d_in[30];
  const float* bnb    = (const float*)d_in[31];
  const float* bnm    = (const float*)d_in[32];
  const float* bnv    = (const float*)d_in[33];
  const float* pw2w   = (const float*)d_in[34];
  const float* pw2b   = (const float*)d_in[35];
  const float* f2lg   = (const float*)d_in[36];
  const float* f2lb   = (const float*)d_in[37];
  const float* f2w1   = (const float*)d_in[38];
  const float* f2b1   = (const float*)d_in[39];
  const float* f2w2   = (const float*)d_in[40];
  const float* f2b2   = (const float*)d_in[41];

  float* out_x    = (float*)d_out;
  float* out_attn = out_x + (size_t)BTD_;

  // ---- workspace carve-out ----
  char* wsb = (char*)d_ws;
  size_t off = 0;
  auto take = [&](size_t bytes) -> char* {
    char* p = wsb + off;
    off += (bytes + 255) & ~(size_t)255;
    return p;
  };
  half_t* hf1w1 = (half_t*)take((size_t)D_ * DFF_ * 2);
  half_t* hf1w2 = (half_t*)take((size_t)DFF_ * D_ * 2);
  half_t* hwq   = (half_t*)take((size_t)D_ * D_ * 2);
  half_t* hwk   = (half_t*)take((size_t)D_ * D_ * 2);
  half_t* hwv   = (half_t*)take((size_t)D_ * D_ * 2);
  half_t* hwpos = (half_t*)take((size_t)D_ * D_ * 2);
  half_t* hwo   = (half_t*)take((size_t)D_ * D_ * 2);
  half_t* hpw1  = (half_t*)take((size_t)D_ * 2 * D_ * 2);
  half_t* hdc   = (half_t*)take((size_t)D_ * D_ * 2);
  half_t* hpw2  = (half_t*)take((size_t)D_ * D_ * 2);
  half_t* hf2w1 = (half_t*)take((size_t)D_ * DFF_ * 2);
  half_t* hf2w2 = (half_t*)take((size_t)DFF_ * D_ * 2);
  half_t* hpe   = (half_t*)take((size_t)T_ * D_ * 2);
  float*  xn_f  = (float*) take((size_t)BTD_ * 4);
  half_t* xn_h  = (half_t*)take((size_t)BTD_ * 2);
  float*  cur   = (float*) take((size_t)BTD_ * 4);
  half_t* hid_h = (half_t*)take((size_t)NT_ * DFF_ * 2);
  float*  qf    = (float*) take((size_t)BTD_ * 4);
  float*  kf32  = (float*) take((size_t)BTD_ * 4);
  float*  vf32  = (float*) take((size_t)BTD_ * 4);
  float*  pfull = (float*) take((size_t)T_ * D_ * 4);
  half_t* qu    = (half_t*)take((size_t)BTD_ * 2);
  half_t* qv    = (half_t*)take((size_t)BTD_ * 2);
  half_t* kh    = (half_t*)take((size_t)BTD_ * 2);
  half_t* vh    = (half_t*)take((size_t)BTD_ * 2);
  half_t* pf    = (half_t*)take((size_t)H_ * T_ * DK_ * 2);
  float*  bdraw = (float*) take((size_t)T_ * T_ * 4);
  half_t* attnh = (half_t*)take((size_t)T_ * T_ * 2);
  float*  obuf  = (float*) take((size_t)BTD_ * 4);
  half_t* oh    = (half_t*)take((size_t)BTD_ * 2);
  float*  c1    = (float*) take((size_t)NT_ * 2 * D_ * 4);
  float*  glu   = (float*) take((size_t)BTD_ * 4);
  half_t* dwh   = (half_t*)take((size_t)BTD_ * 2);
  half_t* dch   = (half_t*)take((size_t)BTD_ * 2);
  float*  bnsc  = (float*) take((size_t)D_ * 4);
  float*  bnsh  = (float*) take((size_t)D_ * 4);

  auto cvt = [&](const float* src, half_t* dst, size_t n) {
    cvt_f32_f16<<<dim3((unsigned)(n / 256)), 256, 0, stream>>>(src, dst, (int)n);
  };

  // ---- weight (and pe) f16 staging ----
  cvt(f1w1, hf1w1, (size_t)D_ * DFF_);
  cvt(f1w2, hf1w2, (size_t)DFF_ * D_);
  cvt(wq,   hwq,   (size_t)D_ * D_);
  cvt(wk,   hwk,   (size_t)D_ * D_);
  cvt(wv,   hwv,   (size_t)D_ * D_);
  cvt(wpos, hwpos, (size_t)D_ * D_);
  cvt(wo,   hwo,   (size_t)D_ * D_);
  cvt(pw1w, hpw1,  (size_t)D_ * 2 * D_);
  cvt(dcw,  hdc,   (size_t)D_ * D_);
  cvt(pw2w, hpw2,  (size_t)D_ * D_);
  cvt(f2w1, hf2w1, (size_t)D_ * DFF_);
  cvt(f2w2, hf2w2, (size_t)DFF_ * D_);
  cvt(pe,   hpe,   (size_t)T_ * D_);

  const dim3 blk(256);
  const dim3 gFFup(DFF_ / BN, NT_ / BM);
  const dim3 gFFdn(D_ / BN, NT_ / BM);
  const dim3 gPW1(2 * D_ / BN, NT_ / BM);
  const dim3 gP(D_ / BN, T_ / BM);
  const dim3 gScore(T_ / BN, T_ / BM);
  const dim3 gAV(DK_ / BN, T_ / BM);

  // ---- macaron FFN 1: cur = x + 0.5*FF(LN(x)) ----
  ln_kernel<<<NT_, blk, 0, stream>>>(x_in, f1lg, f1lb, xn_f, xn_h);
  gemm_nn<true,1,false,false,true><<<gFFup, blk, 0, stream>>>(
      xn_h, D_, hf1w1, DFF_, D_, f1b1, nullptr, nullptr, nullptr, 0.f, nullptr, DFF_, hid_h);
  gemm_nn<true,0,true,true,false><<<gFFdn, blk, 0, stream>>>(
      hid_h, DFF_, hf1w2, D_, DFF_, f1b2, nullptr, nullptr, x_in, 0.5f, cur, D_, nullptr);

  // ---- rel-MHA ----
  ln_kernel<<<NT_, blk, 0, stream>>>(cur, nlg, nlb, xn_f, xn_h);
  gemm_nn<true,0,false,true,false><<<gFFdn, blk, 0, stream>>>(
      xn_h, D_, hwq, D_, D_, bq, nullptr, nullptr, nullptr, 0.f, qf, D_, nullptr);
  gemm_nn<true,0,false,true,false><<<gFFdn, blk, 0, stream>>>(
      xn_h, D_, hwk, D_, D_, bk, nullptr, nullptr, nullptr, 0.f, kf32, D_, nullptr);
  gemm_nn<true,0,false,true,false><<<gFFdn, blk, 0, stream>>>(
      xn_h, D_, hwv, D_, D_, bv, nullptr, nullptr, nullptr, 0.f, vf32, D_, nullptr);
  gemm_nn<false,0,false,true,false><<<gP, blk, 0, stream>>>(
      hpe, D_, hwpos, D_, D_, nullptr, nullptr, nullptr, nullptr, 0.f, pfull, D_, nullptr);
  split_heads<<<BTD_ / 256, blk, 0, stream>>>(qf, kf32, vf32, pos_u, pos_v,
                                              qu, qv, kh, vh);
  split_p<<<(T_ * D_) / 256, blk, 0, stream>>>(pfull, pf);

  for (int bh = 0; bh < B_ * H_; ++bh) {
    const int b = bh / H_, h = bh % H_;
    float*  attn_bh = out_attn + (size_t)bh * T_ * T_;
    const half_t* qu_bh = qu + (size_t)bh * T_ * DK_;
    const half_t* qv_bh = qv + (size_t)bh * T_ * DK_;
    const half_t* kh_bh = kh + (size_t)bh * T_ * DK_;
    const half_t* vh_bh = vh + (size_t)bh * T_ * DK_;
    // ac = (q+u) K^T     -> attn_bh   (async-to-LDS staged WMMA)
    gemm_nt<<<gScore, blk, 0, stream>>>(qu_bh, DK_, kh_bh, DK_, DK_, attn_bh, T_);
    // bd_raw = (q+v) P^T -> ws
    gemm_nt<<<gScore, blk, 0, stream>>>(qv_bh, DK_, pf + (size_t)h * T_ * DK_,
                                        DK_, DK_, bdraw, T_);
    // rel-shift + mask + softmax (f32 attn out + f16 copy for attn@V)
    shift_softmax<<<T_, blk, 0, stream>>>(attn_bh, bdraw, mask + (size_t)b * T_, attnh);
    // o_bh = attn V  (scatter into [B,T,H,DK] via ldc=D)
    gemm_nn<false,0,false,true,false><<<gAV, blk, 0, stream>>>(
        attnh, T_, vh_bh, DK_, T_, nullptr, nullptr, nullptr, nullptr, 0.f,
        obuf + (size_t)b * T_ * D_ + (size_t)h * DK_, D_, nullptr);
  }

  // out proj + residual: cur = cur + (o wo + bo)
  cvt(obuf, oh, (size_t)BTD_);
  gemm_nn<true,0,true,true,false><<<gFFdn, blk, 0, stream>>>(
      oh, D_, hwo, D_, D_, bo, nullptr, nullptr, cur, 1.0f, cur, D_, nullptr);

  // ---- conv module ----
  ln_kernel<<<NT_, blk, 0, stream>>>(cur, cvlg, cvlb, xn_f, xn_h);
  gemm_nn<true,0,false,true,false><<<gPW1, blk, 0, stream>>>(
      xn_h, D_, hpw1, 2 * D_, D_, pw1b, nullptr, nullptr, nullptr, 0.f, c1, 2 * D_, nullptr);
  glu_kernel<<<BTD_ / 256, blk, 0, stream>>>(c1, glu);
  dwconv_kernel<<<BTD_ / 256, blk, 0, stream>>>(glu, dww, dwb, dwh);
  bnprep_kernel<<<1, blk, 0, stream>>>(bng, bnb, bnm, bnv, dcb, bnsc, bnsh);
  gemm_nn<false,2,false,false,true><<<gFFdn, blk, 0, stream>>>(
      dwh, D_, hdc, D_, D_, nullptr, bnsc, bnsh, nullptr, 0.f, nullptr, D_, dch);
  gemm_nn<true,0,true,true,false><<<gFFdn, blk, 0, stream>>>(
      dch, D_, hpw2, D_, D_, pw2b, nullptr, nullptr, cur, 1.0f, cur, D_, nullptr);

  // ---- macaron FFN 2: out_x = cur + FF(LN(cur)) ----
  ln_kernel<<<NT_, blk, 0, stream>>>(cur, f2lg, f2lb, xn_f, xn_h);
  gemm_nn<true,1,false,false,true><<<gFFup, blk, 0, stream>>>(
      xn_h, D_, hf2w1, DFF_, D_, f2b1, nullptr, nullptr, nullptr, 0.f, nullptr, DFF_, hid_h);
  gemm_nn<true,0,true,true,false><<<gFFdn, blk, 0, stream>>>(
      hid_h, DFF_, hf2w2, D_, DFF_, f2b2, nullptr, nullptr, cur, 1.0f, out_x, D_, nullptr);
}
